// DGCNN_14645838479379
// MI455X (gfx1250) — compile-verified, matrix operand order checked by reading, not run on hardware
//
#include <hip/hip_runtime.h>
#include <hip/hip_bf16.h>
#include <stdint.h>

#define K20    20
#define EPSV   1e-5f
#define SLOPEV 0.2f
#define BATCH  8
#define NPTS   2048
#define MROWS  (BATCH*NPTS*K20)   // 327680 edge rows
#define NPOINT (BATCH*NPTS)       // 16384 points

typedef __attribute__((ext_vector_type(16))) _Float16 v16h;
typedef __attribute__((ext_vector_type(8)))  float    v8f;
typedef __attribute__((ext_vector_type(4)))  int      v4i;
typedef __attribute__((ext_vector_type(8)))  int      v8i;

// ---------------------------------------------------------------------------
// 16-bit WMMA operand fragment load (CDNA5 A/B layout, wave32).
// Lane L: row = L&15, half = L>>4. VGPR v holds K pairs:
//   v0..3 -> K = {0..7} + 8*half ; v4..7 -> K = {16..23} + 8*half
// Works for A (row = M of activations) and B (row = output channel of
// row-major weights); row-major, even stride. Also valid on LDS pointers.
// ---------------------------------------------------------------------------
__device__ __forceinline__ v16h load_frag16(const _Float16* base,
                                            int row0, int stride, int kbase,
                                            int lane) {
  int m  = lane & 15;
  int hf = lane >> 4;
  const uint32_t* rp = (const uint32_t*)(base + (size_t)(row0 + m) * stride
                                              + kbase + hf * 8);
  union { v16h h; uint32_t u[8]; } f;
  f.u[0] = rp[0];  f.u[1] = rp[1];  f.u[2] = rp[2];  f.u[3] = rp[3];
  f.u[4] = rp[8];  f.u[5] = rp[9];  f.u[6] = rp[10]; f.u[7] = rp[11];
  return f.h;
}

// ---------------------------------------------------------------------------
// TDM: stage a COUT x CIN f16 weight matrix into LDS (row-major, same layout).
// D# descriptor per CDNA5 ISA 8.3/8.4: 2D tensor, tile == whole tensor.
// Issued by one wave; EXEC is ignored by tensor ops. 2-group form (<=2D).
// ---------------------------------------------------------------------------
template<int CIN, int COUT>
__device__ __forceinline__ void tdm_load_weights(const _Float16* W,
                                                 _Float16* wlds) {
  uint64_t ga  = (uint64_t)(uintptr_t)W;
  uint32_t lds = (uint32_t)(uintptr_t)wlds;     // flat addr[31:0] = LDS offset
  v4i g0;
  g0[0] = 1;                                    // count=1, user mode
  g0[1] = (int)lds;                             // lds_addr [63:32]
  g0[2] = (int)(uint32_t)ga;                    // global_addr [95:64]
  g0[3] = (int)((uint32_t)((ga >> 32) & 0x01FFFFFFu) | (2u << 30)); // type=2
  const uint32_t td0 = CIN, td1 = COUT;
  const uint64_t s0  = CIN;                     // tensor_dim0_stride (elems)
  const uint64_t s1  = (uint64_t)CIN * COUT;    // tensor_dim1_stride (unused 2D)
  v8i g1;
  g1[0] = (int)(1u << 16);                      // wg_mask=0, data_size=1 (2B)
  g1[1] = (int)((td0 & 0xFFFFu) << 16);         // tensor_dim0 [79:48]
  g1[2] = (int)(((td0 >> 16) & 0xFFFFu) | ((td1 & 0xFFFFu) << 16));
  g1[3] = (int)(((td1 >> 16) & 0xFFFFu) | ((uint32_t)CIN << 16)); // tile_dim0
  g1[4] = (int)((uint32_t)COUT & 0xFFFFu);      // tile_dim1, tile_dim2=0
  g1[5] = (int)(uint32_t)s0;                    // stride0 [207:160]
  g1[6] = (int)(((uint32_t)(s0 >> 32) & 0xFFFFu) | (((uint32_t)s1 & 0xFFFFu) << 16));
  g1[7] = (int)(uint32_t)(s1 >> 16);
  asm volatile("tensor_load_to_lds %0, %1" :: "s"(g0), "s"(g1) : "memory");
  __builtin_amdgcn_s_wait_tensorcnt(0);
}

// Exact (order-independent) float atomic max via sign-split integer atomics.
__device__ __forceinline__ void atomicMaxF(float* addr, float val) {
  if (val >= 0.0f) atomicMax((int*)addr, __float_as_int(val));
  else             atomicMin((unsigned int*)addr, __float_as_uint(val));
}

// ---------------------------------------------------------------------------
// KNN: per batch, stage points + squared norms in LDS, thread-per-point
// stable 20-deep insertion using d = 2*dot - |xi|^2 - |xj|^2 (matches ref).
// ---------------------------------------------------------------------------
__global__ void __launch_bounds__(256)
knn_k(const float* __restrict__ x, int* __restrict__ idx) {
  __shared__ float xs0[NPTS], xs1[NPTS], xs2[NPTS], xq[NPTS];
  int b  = blockIdx.x >> 3;                       // 8 blocks per batch
  int ib = (blockIdx.x & 7) * 256 + threadIdx.x;  // point index
  const float* xb = x + (size_t)b * 3 * NPTS;
  for (int t = threadIdx.x; t < NPTS; t += 256) {
    float a0 = xb[t], a1 = xb[NPTS + t], a2 = xb[2 * NPTS + t];
    xs0[t] = a0; xs1[t] = a1; xs2[t] = a2;
    xq[t] = a0 * a0 + a1 * a1 + a2 * a2;
  }
  __syncthreads();
  float xi0 = xs0[ib], xi1 = xs1[ib], xi2 = xs2[ib], xxi = xq[ib];
  float best[K20]; int besti[K20];
#pragma unroll
  for (int s = 0; s < K20; ++s) { best[s] = -__builtin_inff(); besti[s] = 0; }
  for (int j = 0; j < NPTS; ++j) {
    float d = 2.0f * (xi0 * xs0[j] + xi1 * xs1[j] + xi2 * xs2[j]) - xxi - xq[j];
    if (d > best[K20 - 1]) {
      float v = d; int vi = j;
#pragma unroll
      for (int s = 0; s < K20; ++s) {           // stable insertion (ties keep
        bool gt = v > best[s];                  // lower index, like top_k)
        float tv = best[s]; int ti = besti[s];
        if (gt) { best[s] = v; besti[s] = vi; v = tv; vi = ti; }
      }
    }
  }
  int* op = idx + ((size_t)b * NPTS + ib) * K20;
#pragma unroll
  for (int s = 0; s < K20; ++s) op[s] = besti[s];
}

// ---------------------------------------------------------------------------
// Edge features: [xj-xi (3), xi (3)] zero-padded to 32 f16 cols for WMMA K=32.
// ---------------------------------------------------------------------------
__global__ void edge_k(const float* __restrict__ x, const int* __restrict__ idx,
                       _Float16* __restrict__ h0) {
  int m = blockIdx.x * blockDim.x + threadIdx.x;
  if (m >= MROWS) return;
  int b   = m / (NPTS * K20);
  int rem = m - b * NPTS * K20;
  int np  = rem / K20;
  int ke  = rem - np * K20;
  int j   = idx[((size_t)b * NPTS + np) * K20 + ke];
  const float* xb = x + (size_t)b * 3 * NPTS;
  float xi0 = xb[np], xi1 = xb[NPTS + np], xi2 = xb[2 * NPTS + np];
  float xj0 = xb[j],  xj1 = xb[NPTS + j],  xj2 = xb[2 * NPTS + j];
  float v[6] = {xj0 - xi0, xj1 - xi1, xj2 - xi2, xi0, xi1, xi2};
  _Float16* row = h0 + (size_t)m * 32;
#pragma unroll
  for (int t = 0; t < 6; ++t)  row[t] = (_Float16)v[t];
#pragma unroll
  for (int t = 6; t < 32; ++t) row[t] = (_Float16)0.0f;
}

// fp32 -> f16 weight convert with optional K zero-padding (row-major)
__global__ void cvtpad_k(const float* __restrict__ src, _Float16* __restrict__ dst,
                         int rows, int cin, int cinp) {
  int t = blockIdx.x * blockDim.x + threadIdx.x;
  if (t >= rows * cinp) return;
  int r = t / cinp, c = t - r * cinp;
  dst[t] = (c < cin) ? (_Float16)src[r * cin + c] : (_Float16)0.0f;
}

__global__ void fillneg_k(float* __restrict__ p, int n) {
  int t = blockIdx.x * blockDim.x + threadIdx.x;
  if (t < n) p[t] = -__builtin_inff();
}

__global__ void cvt_f32_f16_k(const float* __restrict__ s, _Float16* __restrict__ d, int n) {
  int t = blockIdx.x * blockDim.x + threadIdx.x;
  if (t < n) d[t] = (_Float16)s[t];
}

// ---------------------------------------------------------------------------
// Pass 1: WMMA GEMM + deterministic per-block channel sum / sumsq partials.
// Wave owns one 16-row M tile. Weights staged in LDS via TDM when they fit;
// A fragments cached in registers for CIN <= 128.
// ---------------------------------------------------------------------------
template<int CIN, int COUT>
__global__ void __launch_bounds__(128)
gemm_stats_k(const _Float16* __restrict__ A, const _Float16* __restrict__ W,
             float* __restrict__ partials) {
  constexpr bool USE_TDM = ((size_t)CIN * COUT * 2 <= 65536);
  constexpr bool CACHE_A = (CIN <= 128);
  __shared__ _Float16 wlds[USE_TDM ? (CIN * COUT) : 1];
  __shared__ float sred[4][COUT][2];
  int lane = threadIdx.x & 31;
  int wv   = threadIdx.x >> 5;
  int row0 = (blockIdx.x * 4 + wv) * 16;
  if constexpr (USE_TDM) {
    if (threadIdx.x < 32) tdm_load_weights<CIN, COUT>(W, wlds);
    __syncthreads();
  }
  v16h afr[CACHE_A ? CIN / 32 : 1];
  if constexpr (CACHE_A) {
#pragma unroll
    for (int kc = 0; kc < CIN / 32; ++kc)
      afr[kc] = load_frag16(A, row0, CIN, kc * 32, lane);
  }
  for (int nt = 0; nt < COUT / 16; ++nt) {
    v8f acc = {0.f, 0.f, 0.f, 0.f, 0.f, 0.f, 0.f, 0.f};
#pragma unroll
    for (int kc = 0; kc < CIN / 32; ++kc) {
      v16h a, b;
      if constexpr (CACHE_A) a = afr[kc];
      else                   a = load_frag16(A, row0, CIN, kc * 32, lane);
      if constexpr (USE_TDM) b = load_frag16(wlds, nt * 16, CIN, kc * 32, lane);
      else                   b = load_frag16(W,    nt * 16, CIN, kc * 32, lane);
      acc = __builtin_amdgcn_wmma_f32_16x16x32_f16(false, a, false, b,
                                                   (short)0, acc, false, false);
    }
    float s = 0.f, q = 0.f;
#pragma unroll
    for (int r = 0; r < 8; ++r) { s += acc[r]; q += acc[r] * acc[r]; }
    s += __shfl_xor(s, 16);      // combine the two row-halves of each column
    q += __shfl_xor(q, 16);
    if (lane < 16) {
      sred[wv][nt * 16 + lane][0] = s;
      sred[wv][nt * 16 + lane][1] = q;
    }
  }
  __syncthreads();
  for (int c = threadIdx.x; c < COUT; c += 128) {   // fixed-order: deterministic
    float ts = 0.f, tq = 0.f;
#pragma unroll
    for (int w2 = 0; w2 < 4; ++w2) { ts += sred[w2][c][0]; tq += sred[w2][c][1]; }
    size_t o = ((size_t)blockIdx.x * COUT + c) * 2;
    partials[o] = ts; partials[o + 1] = tq;
  }
}

// Fold BN into per-channel scale/shift (fixed-order block sum: deterministic).
__global__ void finalize_k(const float* __restrict__ partials, int nblocks, int cout,
                           const float* __restrict__ g, const float* __restrict__ bb,
                           float* __restrict__ scale, float* __restrict__ shift,
                           float invM) {
  for (int c = threadIdx.x; c < cout; c += blockDim.x) {
    float s = 0.f, q = 0.f;
    for (int bk = 0; bk < nblocks; ++bk) {
      size_t o = ((size_t)bk * cout + c) * 2;
      s += partials[o]; q += partials[o + 1];
    }
    float mu  = s * invM;
    float var = q * invM - mu * mu;
    float sc  = g[c] * rsqrtf(var + EPSV);
    scale[c] = sc;
    shift[c] = bb[c] - mu * sc;
  }
}

// ---------------------------------------------------------------------------
// Pass 2: recompute WMMA GEMM, apply BN affine + LeakyReLU, emit f16
// activations for the next layer, fuse max-over-k into `cat` via atomic max.
// FINAL variant writes the BN+LReLU result transposed to d_out [B,1024,N].
// ---------------------------------------------------------------------------
template<int CIN, int COUT, bool WRITE_H, bool FINAL>
__global__ void __launch_bounds__(128)
gemm_apply_k(const _Float16* __restrict__ A, const _Float16* __restrict__ W,
             const float* __restrict__ scale, const float* __restrict__ shift,
             _Float16* __restrict__ Hout, float* __restrict__ cat, int catoff,
             float* __restrict__ out) {
  constexpr bool USE_TDM = ((size_t)CIN * COUT * 2 <= 65536);
  constexpr bool CACHE_A = (CIN <= 128);
  __shared__ _Float16 wlds[USE_TDM ? (CIN * COUT) : 1];
  int lane = threadIdx.x & 31;
  int wv   = threadIdx.x >> 5;
  int row0 = (blockIdx.x * 4 + wv) * 16;
  int hf   = lane >> 4;
  int nlo  = lane & 15;
  if constexpr (USE_TDM) {
    if (threadIdx.x < 32) tdm_load_weights<CIN, COUT>(W, wlds);
    __syncthreads();
  }
  v16h afr[CACHE_A ? CIN / 32 : 1];
  if constexpr (CACHE_A) {
#pragma unroll
    for (int kc = 0; kc < CIN / 32; ++kc)
      afr[kc] = load_frag16(A, row0, CIN, kc * 32, lane);
  }
  for (int nt = 0; nt < COUT / 16; ++nt) {
    v8f acc = {0.f, 0.f, 0.f, 0.f, 0.f, 0.f, 0.f, 0.f};
#pragma unroll
    for (int kc = 0; kc < CIN / 32; ++kc) {
      v16h a, b;
      if constexpr (CACHE_A) a = afr[kc];
      else                   a = load_frag16(A, row0, CIN, kc * 32, lane);
      if constexpr (USE_TDM) b = load_frag16(wlds, nt * 16, CIN, kc * 32, lane);
      else                   b = load_frag16(W,    nt * 16, CIN, kc * 32, lane);
      acc = __builtin_amdgcn_wmma_f32_16x16x32_f16(false, a, false, b,
                                                   (short)0, acc, false, false);
    }
    int col = nt * 16 + nlo;
    float sc = scale[col], sh = shift[col];
#pragma unroll
    for (int r = 0; r < 8; ++r) {                 // D: VGPR r = row r + 8*half
      int m = row0 + r + 8 * hf;
      float y = acc[r] * sc + sh;
      y = (y >= 0.f) ? y : SLOPEV * y;
      if (FINAL) {
        int bb = m >> 11, np = m & 2047;          // m = b*2048 + n
        out[((size_t)bb * 1024 + col) * 2048 + np] = y;
      } else {
        if (WRITE_H) Hout[(size_t)m * COUT + col] = (_Float16)y;
        int p = m / K20;                          // point id: max over k=20
        atomicMaxF(&cat[(size_t)p * 512 + catoff + col], y);
      }
    }
  }
}

// ---------------------------------------------------------------------------
extern "C" void kernel_launch(void* const* d_in, const int* in_sizes, int n_in,
                              void* d_out, int out_size, void* d_ws, size_t ws_size,
                              hipStream_t stream) {
  (void)in_sizes; (void)n_in; (void)out_size; (void)ws_size;
  const float* x  = (const float*)d_in[0];
  const float* W1 = (const float*)d_in[1];  const float* g1 = (const float*)d_in[2];  const float* b1 = (const float*)d_in[3];
  const float* W2 = (const float*)d_in[4];  const float* g2 = (const float*)d_in[5];  const float* b2 = (const float*)d_in[6];
  const float* W3 = (const float*)d_in[7];  const float* g3 = (const float*)d_in[8];  const float* b3 = (const float*)d_in[9];
  const float* W4 = (const float*)d_in[10]; const float* g4 = (const float*)d_in[11]; const float* b4 = (const float*)d_in[12];
  const float* W5 = (const float*)d_in[13]; const float* g5 = (const float*)d_in[14]; const float* b5 = (const float*)d_in[15];
  float* out = (float*)d_out;

  // ---- workspace layout (256B aligned) ----
  char* ws = (char*)d_ws;
  size_t off = 0;
  auto alloc = [&](size_t bytes) { char* p = ws + off; off = (off + bytes + 255) & ~(size_t)255; return p; };
  int*       idx  = (int*)      alloc((size_t)NPOINT * K20 * 4);
  _Float16*  h0   = (_Float16*) alloc((size_t)MROWS * 32 * 2);
  _Float16*  h1   = (_Float16*) alloc((size_t)MROWS * 64 * 2);
  _Float16*  h2   = (_Float16*) alloc((size_t)MROWS * 64 * 2);
  _Float16*  h3   = (_Float16*) alloc((size_t)MROWS * 128 * 2);
  float*     catf = (float*)    alloc((size_t)NPOINT * 512 * 4);
  _Float16*  cath = (_Float16*) alloc((size_t)NPOINT * 512 * 2);
  _Float16*  w1h  = (_Float16*) alloc(64 * 32 * 2);
  _Float16*  w2h  = (_Float16*) alloc(64 * 64 * 2);
  _Float16*  w3h  = (_Float16*) alloc(128 * 64 * 2);
  _Float16*  w4h  = (_Float16*) alloc(256 * 128 * 2);
  _Float16*  w5h  = (_Float16*) alloc(1024 * 512 * 2);
  float*     part = (float*)    alloc((size_t)5120 * 256 * 2 * 4); // max partials
  float*     scb  = (float*)    alloc(1024 * 4);
  float*     shb  = (float*)    alloc(1024 * 4);

  const int TPB = 256;
  const int MT_BLK = (MROWS / 16) / 4;   // 5120 GEMM blocks (4 waves each)
  const int PT_BLK = (NPOINT / 16) / 4;  // 256 blocks for the final GEMM
  const float invM  = 1.0f / (float)MROWS;
  const float invMP = 1.0f / (float)NPOINT;

  // weights -> f16 (W1 zero-padded 6 -> 32)
  cvtpad_k<<<(64 * 32 + TPB - 1) / TPB, TPB, 0, stream>>>(W1, w1h, 64, 6, 32);
  cvtpad_k<<<(64 * 64 + TPB - 1) / TPB, TPB, 0, stream>>>(W2, w2h, 64, 64, 64);
  cvtpad_k<<<(128 * 64 + TPB - 1) / TPB, TPB, 0, stream>>>(W3, w3h, 128, 64, 64);
  cvtpad_k<<<(256 * 128 + TPB - 1) / TPB, TPB, 0, stream>>>(W4, w4h, 256, 128, 128);
  cvtpad_k<<<(1024 * 512 + TPB - 1) / TPB, TPB, 0, stream>>>(W5, w5h, 1024, 512, 512);
  fillneg_k<<<(NPOINT * 512 + TPB - 1) / TPB, TPB, 0, stream>>>(catf, NPOINT * 512);

  knn_k<<<BATCH * (NPTS / 256), 256, 0, stream>>>(x, idx);
  edge_k<<<(MROWS + TPB - 1) / TPB, TPB, 0, stream>>>(x, idx, h0);

  // conv1: 32(pad) -> 64
  gemm_stats_k<32, 64><<<MT_BLK, 128, 0, stream>>>(h0, w1h, part);
  finalize_k<<<1, 256, 0, stream>>>(part, MT_BLK, 64, g1, b1, scb, shb, invM);
  gemm_apply_k<32, 64, true, false><<<MT_BLK, 128, 0, stream>>>(h0, w1h, scb, shb, h1, catf, 0, nullptr);

  // conv2: 64 -> 64
  gemm_stats_k<64, 64><<<MT_BLK, 128, 0, stream>>>(h1, w2h, part);
  finalize_k<<<1, 256, 0, stream>>>(part, MT_BLK, 64, g2, b2, scb, shb, invM);
  gemm_apply_k<64, 64, true, false><<<MT_BLK, 128, 0, stream>>>(h1, w2h, scb, shb, h2, catf, 64, nullptr);

  // conv3: 64 -> 128
  gemm_stats_k<64, 128><<<MT_BLK, 128, 0, stream>>>(h2, w3h, part);
  finalize_k<<<1, 256, 0, stream>>>(part, MT_BLK, 128, g3, b3, scb, shb, invM);
  gemm_apply_k<64, 128, true, false><<<MT_BLK, 128, 0, stream>>>(h2, w3h, scb, shb, h3, catf, 128, nullptr);

  // conv4: 128 -> 256 (only max-over-k needed)
  gemm_stats_k<128, 256><<<MT_BLK, 128, 0, stream>>>(h3, w4h, part);
  finalize_k<<<1, 256, 0, stream>>>(part, MT_BLK, 256, g4, b4, scb, shb, invM);
  gemm_apply_k<128, 256, false, false><<<MT_BLK, 128, 0, stream>>>(h3, w4h, scb, shb, nullptr, catf, 256, nullptr);

  // cat fp32 -> f16, final GEMM 512 -> 1024 with BN over (B,N), transposed out
  cvt_f32_f16_k<<<(NPOINT * 512 + TPB - 1) / TPB, TPB, 0, stream>>>(catf, cath, NPOINT * 512);
  gemm_stats_k<512, 1024><<<PT_BLK, 128, 0, stream>>>(cath, w5h, part);
  finalize_k<<<1, 256, 0, stream>>>(part, PT_BLK, 1024, g5, b5, scb, shb, invMP);
  gemm_apply_k<512, 1024, false, true><<<PT_BLK, 128, 0, stream>>>(cath, w5h, scb, shb, nullptr, nullptr, 0, out);
}
